// CLI_v3_63702954934487
// MI455X (gfx1250) — compile-verified, matrix operand order checked by reading, not run on hardware
//
#include <hip/hip_runtime.h>

typedef __bf16 bf16_t;
typedef __attribute__((ext_vector_type(16))) __bf16 v16bf;
typedef __attribute__((ext_vector_type(8)))  __bf16 v8bf;
typedef __attribute__((ext_vector_type(8)))  float  v8f;

#define NA      8192
#define NB      8192
#define DD      256
#define NBATCH  4
#define KTOP    8

// ---------------------------------------------------------------------------
// Kernel 1: repack w1^T (512x256) and w2^T (256x256) into bf16 WMMA B-fragment
// layout: tile t = kt*16+nt, per tile 32 lanes x 16 bf16 contiguous (32B/lane).
// Per-lane element e maps to klocal = 8*(lane/16) + e (e<8) or 16+8*(lane/16)+(e-8),
// n = nt*16 + lane%16  (mirrors the 16-bit A/B VGPR striping in CDNA5 ISA).
// ---------------------------------------------------------------------------
__global__ __launch_bounds__(256) void cli_convert_weights(
    const float* __restrict__ w1, const float* __restrict__ w2,
    bf16_t* __restrict__ w1fr, bf16_t* __restrict__ w2fr)
{
    const int o = blockIdx.x * 256 + threadIdx.x;
    const int W1E = DD * 2 * DD;   // 131072
    const int W2E = DD * DD;       // 65536
    if (o < W1E) {
        int t = o >> 9, r = o & 511;
        int L = r >> 4, e = r & 15;
        int kt = t >> 4, nt = t & 15;
        int klocal = (e < 8) ? (8 * (L >> 4) + e) : (16 + 8 * (L >> 4) + (e - 8));
        int k = kt * 32 + klocal;
        int n = nt * 16 + (L & 15);
        w1fr[o] = (bf16_t)w1[n * (2 * DD) + k];   // B[k][n] = w1[n][k]
    } else if (o < W1E + W2E) {
        int o2 = o - W1E;
        int t = o2 >> 9, r = o2 & 511;
        int L = r >> 4, e = r & 15;
        int kt = t >> 4, nt = t & 15;
        int klocal = (e < 8) ? (8 * (L >> 4) + e) : (16 + 8 * (L >> 4) + (e - 8));
        int k = kt * 32 + klocal;
        int n = nt * 16 + (L & 15);
        w2fr[o2] = (bf16_t)w2[n * DD + k];        // B[k][n] = w2[n][k]
    }
}

// ---------------------------------------------------------------------------
// Kernel 2: per a-point brute-force top-8 nearest b-points on (coords//16).
// coords_b tiles staged through LDS (2048 pts = 24KB), register insertion sort.
// Emits neighbor index + dw = R - clip(dist/128, 0, R).
// ---------------------------------------------------------------------------
__global__ __launch_bounds__(256) void cli_topk(
    const int* __restrict__ coords_a, const int* __restrict__ coords_b,
    int* __restrict__ topk_idx, float* __restrict__ topk_dw)
{
    const int b   = blockIdx.y;
    const int tid = threadIdx.x;
    const int a   = blockIdx.x * 256 + tid;

    __shared__ float scb[2048 * 3];

    const float cax = (float)(coords_a[((size_t)b * NA + a) * 3 + 0] >> 4);
    const float cay = (float)(coords_a[((size_t)b * NA + a) * 3 + 1] >> 4);
    const float caz = (float)(coords_a[((size_t)b * NA + a) * 3 + 2] >> 4);

    float bd[KTOP];
    int   bi[KTOP];
#pragma unroll
    for (int i = 0; i < KTOP; ++i) { bd[i] = 3.0e38f; bi[i] = 0; }

    for (int ch = 0; ch < NB; ch += 2048) {
        __syncthreads();
        for (int e = tid; e < 2048 * 3; e += 256)
            scb[e] = (float)(coords_b[((size_t)b * NB + ch) * 3 + e] >> 4);
        __syncthreads();
        for (int j = 0; j < 2048; ++j) {
            float dx = cax - scb[j * 3 + 0];
            float dy = cay - scb[j * 3 + 1];
            float dz = caz - scb[j * 3 + 2];
            float d2 = dx * dx + dy * dy + dz * dz;
            if (d2 < bd[KTOP - 1]) {                 // usually false: cheap reject
                bd[KTOP - 1] = d2; bi[KTOP - 1] = ch + j;
#pragma unroll
                for (int p = KTOP - 1; p > 0; --p) { // one bubble pass re-sorts
                    if (bd[p] < bd[p - 1]) {
                        float td = bd[p]; bd[p] = bd[p - 1]; bd[p - 1] = td;
                        int   ti = bi[p]; bi[p] = bi[p - 1]; bi[p - 1] = ti;
                    }
                }
            }
        }
    }
#pragma unroll
    for (int k = 0; k < KTOP; ++k) {
        topk_idx[((size_t)(b * NA + a)) * KTOP + k] = bi[k];
        float dist = sqrtf(bd[k]) * (1.0f / 128.0f);
        topk_dw[((size_t)(b * NA + a)) * KTOP + k] =
            0.5f - fminf(fmaxf(dist, 0.0f), 0.5f);
    }
}

// ---------------------------------------------------------------------------
// Kernel 3: fused gather + 2-layer MLP + k-reduction using bf16 WMMA.
// One block = 16-row tile (2 a-points x 8 neighbors). 8 waves x 2 N-tiles.
// ---------------------------------------------------------------------------
__global__ __launch_bounds__(256) void cli_mlp_wmma(
    const float* __restrict__ a_feats, const float* __restrict__ b_feats,
    const int* __restrict__ topk_idx, const float* __restrict__ topk_dw,
    const bf16_t* __restrict__ w1fr, const float* __restrict__ b1,
    const bf16_t* __restrict__ w2fr, const float* __restrict__ b2,
    float* __restrict__ out)
{
    const int b    = blockIdx.y;
    const int a0   = blockIdx.x * 2;
    const int tid  = threadIdx.x;
    const int wave = tid >> 5;
    const int lane = tid & 31;

    // Padded rows: 520*2B = 1040B and 264*2B = 528B strides -> bank-staggered
    // ds_load_b128 A-fragment reads across the 16 M-lanes.
    __shared__ __align__(16) bf16_t sX[16][520];   // X = [nb | af-nb], 16x512
    __shared__ __align__(16) bf16_t sH[16][264];   // H after relu*dw,  16x256
    __shared__ float sDW[16];
    __shared__ int   sIdx[16];

    if (tid < 16) {
        int a = a0 + (tid >> 3);
        int k = tid & 7;
        sIdx[tid] = topk_idx[((size_t)(b * NA + a)) * KTOP + k];
        sDW[tid]  = topk_dw [((size_t)(b * NA + a)) * KTOP + k];
    }
    __syncthreads();

    // Gather + build X in bf16 (dominant HBM traffic of this kernel).
    for (int e = tid; e < 16 * 512; e += 256) {
        int r = e >> 9, c = e & 511;
        int a   = a0 + (r >> 3);
        int nbi = sIdx[r];
        float v;
        if (c < DD) {
            v = b_feats[((size_t)b * NB + nbi) * DD + c];
        } else {
            int cc = c - DD;
            v = a_feats[((size_t)b * NA + a) * DD + cc]
              - b_feats[((size_t)b * NB + nbi) * DD + cc];
        }
        sX[r][c] = (bf16_t)v;
    }
    // Passthrough: out[..., 0:256] = a_feats
    for (int e = tid; e < 512; e += 256) {
        int a = a0 + (e >> 8);
        int c = e & 255;
        out[((size_t)b * NA + a) * (2 * DD) + c] =
            a_feats[((size_t)b * NA + a) * DD + c];
    }
    __syncthreads();

    const int nt0   = wave;        // N-tiles 0..7
    const int nt1   = wave + 8;    // N-tiles 8..15
    const int m     = lane & 15;   // A-fragment row
    const int khalf = (lane >> 4) * 8;

    // ---------------- GEMM1: H = relu(X * W1^T + b1) * dw,  K = 512 ---------
    v8f acc0 = {}; v8f acc1 = {};
#pragma unroll 4
    for (int kt = 0; kt < 16; ++kt) {
        int kb = kt * 32 + khalf;
        v8bf lo = *(const v8bf*)&sX[m][kb];
        v8bf hi = *(const v8bf*)&sX[m][kb + 16];
        v16bf A;
#pragma unroll
        for (int i = 0; i < 8; ++i) { A[i] = lo[i]; A[i + 8] = hi[i]; }
        v16bf B0 = *(const v16bf*)(w1fr + (((size_t)(kt * 16 + nt0)) * 32 + lane) * 16);
        v16bf B1 = *(const v16bf*)(w1fr + (((size_t)(kt * 16 + nt1)) * 32 + lane) * 16);
        acc0 = __builtin_amdgcn_wmma_f32_16x16x32_bf16(
                   false, A, false, B0, (short)0, acc0, false, false);
        acc1 = __builtin_amdgcn_wmma_f32_16x16x32_bf16(
                   false, A, false, B1, (short)0, acc1, false, false);
    }
    {   // epilogue: C layout -> lane<16: M=j, N=lane; lane>=16: M=j+8, N=lane-16
        int nc0 = nt0 * 16 + (lane & 15);
        int nc1 = nt1 * 16 + (lane & 15);
        float bia0 = b1[nc0], bia1 = b1[nc1];
#pragma unroll
        for (int j = 0; j < 8; ++j) {
            int mr    = j + ((lane >> 4) << 3);
            float dwv = sDW[mr];
            sH[mr][nc0] = (bf16_t)(fmaxf(acc0[j] + bia0, 0.0f) * dwv);
            sH[mr][nc1] = (bf16_t)(fmaxf(acc1[j] + bia1, 0.0f) * dwv);
        }
    }
    __syncthreads();

    // ---------------- GEMM2: O = H * W2^T, K = 256, then sum over k ---------
    v8f o0 = {}; v8f o1 = {};
#pragma unroll
    for (int kt = 0; kt < 8; ++kt) {
        int kb = kt * 32 + khalf;
        v8bf lo = *(const v8bf*)&sH[m][kb];
        v8bf hi = *(const v8bf*)&sH[m][kb + 16];
        v16bf A;
#pragma unroll
        for (int i = 0; i < 8; ++i) { A[i] = lo[i]; A[i + 8] = hi[i]; }
        v16bf B0 = *(const v16bf*)(w2fr + (((size_t)(kt * 16 + nt0)) * 32 + lane) * 16);
        v16bf B1 = *(const v16bf*)(w2fr + (((size_t)(kt * 16 + nt1)) * 32 + lane) * 16);
        o0 = __builtin_amdgcn_wmma_f32_16x16x32_bf16(
                 false, A, false, B0, (short)0, o0, false, false);
        o1 = __builtin_amdgcn_wmma_f32_16x16x32_bf16(
                 false, A, false, B1, (short)0, o1, false, false);
    }
    // Row reduction = sum over k neighbors: lanes 0-15 hold rows 0-7 (a-point 0),
    // lanes 16-31 hold rows 8-15 (a-point 1). b2 added once per k => 8*b2.
    float s0 = 0.0f, s1 = 0.0f;
#pragma unroll
    for (int j = 0; j < 8; ++j) { s0 += o0[j]; s1 += o1[j]; }
    int nc0 = nt0 * 16 + (lane & 15);
    int nc1 = nt1 * 16 + (lane & 15);
    s0 += 8.0f * b2[nc0];
    s1 += 8.0f * b2[nc1];
    int ap = a0 + (lane >> 4);
    out[((size_t)b * NA + ap) * (2 * DD) + DD + nc0] = s0;
    out[((size_t)b * NA + ap) * (2 * DD) + DD + nc1] = s1;
}

// ---------------------------------------------------------------------------
extern "C" void kernel_launch(void* const* d_in, const int* in_sizes, int n_in,
                              void* d_out, int out_size, void* d_ws, size_t ws_size,
                              hipStream_t stream)
{
    (void)in_sizes; (void)n_in; (void)out_size; (void)ws_size;
    const float* a_feats  = (const float*)d_in[0];
    const float* b_feats  = (const float*)d_in[1];
    const int*   coords_a = (const int*)  d_in[2];
    const int*   coords_b = (const int*)  d_in[3];
    const float* w1       = (const float*)d_in[4];
    const float* b1       = (const float*)d_in[5];
    const float* w2       = (const float*)d_in[6];
    const float* b2       = (const float*)d_in[7];
    float* out = (float*)d_out;

    char* ws = (char*)d_ws;
    int*    topk_idx = (int*)   ws;                               // 1 MB
    float*  topk_dw  = (float*) (ws + (1u << 20));                // 1 MB
    bf16_t* w1fr     = (bf16_t*)(ws + (2u << 20));                // 256 KB
    bf16_t* w2fr     = (bf16_t*)(ws + (2u << 20) + (256u << 10)); // 128 KB

    cli_convert_weights<<<dim3((131072 + 65536) / 256), dim3(256), 0, stream>>>(
        w1, w2, w1fr, w2fr);
    cli_topk<<<dim3(NA / 256, NBATCH), dim3(256), 0, stream>>>(
        coords_a, coords_b, topk_idx, topk_dw);
    cli_mlp_wmma<<<dim3(NA / 2, NBATCH), dim3(256), 0, stream>>>(
        a_feats, b_feats, topk_idx, topk_dw, w1fr, b1, w2fr, b2, out);
}
